// VectorQuantizer_51917564674215
// MI455X (gfx1250) — matrix operand order repaired and checked by reading, not compiled, not run on hardware
//
#include <hip/hip_runtime.h>

#define B_ROWS  32768
#define K_CODES 8192
#define D_DIM   512

typedef __attribute__((ext_vector_type(16))) _Float16 v16h;
typedef __attribute__((ext_vector_type(8)))  _Float16 v8h;
typedef __attribute__((ext_vector_type(4)))  _Float16 v4h;
typedef __attribute__((ext_vector_type(8)))  float    v8f;

// -------- Kernel 1: codebook f32 -> f16 + per-code sum of squares ----------
__global__ __launch_bounds__(128) void vq_prep_codebook(const float* __restrict__ cb,
                                                        _Float16* __restrict__ cbh,
                                                        float* __restrict__ csq) {
    __shared__ float red[128];
    const int row = blockIdx.x;
    const float* src = cb + (size_t)row * D_DIM;
    _Float16* dst = cbh + (size_t)row * D_DIM;

    const int i = threadIdx.x * 4;   // 128 threads * 4 = 512 exactly
    float4 f = *(const float4*)(src + i);
    v4h h = { (_Float16)f.x, (_Float16)f.y, (_Float16)f.z, (_Float16)f.w };
    *(v4h*)(dst + i) = h;
    float s = f.x * f.x + f.y * f.y + f.z * f.z + f.w * f.w;

    red[threadIdx.x] = s;
    __syncthreads();
    for (int off = 64; off > 0; off >>= 1) {
        if (threadIdx.x < off) red[threadIdx.x] += red[threadIdx.x + off];
        __syncthreads();
    }
    if (threadIdx.x == 0) csq[row] = red[0];
}

// Load one 32x16 B fragment (code tile) from the padded LDS tile.
__device__ __forceinline__ v16h load_bfrag(const _Float16* tp, int lh, int hi, int kk) {
    const _Float16* bp = tp + lh * 520 + kk * 32 + hi * 8;
    v8h b0 = *(const v8h*)bp;
    v8h b1 = *(const v8h*)(bp + 16);
    return __builtin_shufflevector(b0, b1,
               0, 1, 2, 3, 4, 5, 6, 7, 8, 9, 10, 11, 12, 13, 14, 15);
}

// -------- Kernel 2: WMMA distance GEMM + running argmin --------------------
// 4 waves per WG; each wave owns 16 rows; WG covers 64 rows.
// Double-buffered LDS tiles with register-staged prefetch; B-fragments are
// explicitly software-pipelined one stage ahead so the ds_load for WMMA k+1
// is in flight while WMMA k executes (forces two live B register buffers).
__global__ __launch_bounds__(128) void vq_argmin(const float* __restrict__ z,
                                                 const _Float16* __restrict__ cbh,
                                                 const float* __restrict__ csq,
                                                 int* __restrict__ idx_i,
                                                 float* __restrict__ idx_f) {
    // 16 codes x 512 halves, padded row stride 520 halves (1040B) for
    // conflict-free ds_load_b128 across the 16-lane groups. Two buffers.
    __shared__ _Float16 tile[2][16 * 520];

    const int wave = threadIdx.x >> 5;
    const int lane = threadIdx.x & 31;
    const int lh   = lane & 15;      // position within 16-lane group
    const int hi   = lane >> 4;      // 0 = lanes 0-15, 1 = lanes 16-31
    const int r0   = blockIdx.x * 64 + wave * 16;

    // ---- Build A fragments (16x32 f16 per WMMA, 16 k-chunks of D=512). ----
    // ISA layout: lanes 0-15 row M=lh hold K {0..7, 16..23};
    //             lanes 16-31 row M=lh hold K {8..15, 24..31}.
    v16h A[16];
    {
        const float* zr = z + (size_t)(r0 + lh) * D_DIM;
        #pragma unroll
        for (int kk = 0; kk < 16; ++kk) {
            const int kb = kk * 32 + hi * 8;
            float4 f0 = *(const float4*)(zr + kb);
            float4 f1 = *(const float4*)(zr + kb + 4);
            float4 f2 = *(const float4*)(zr + kb + 16);
            float4 f3 = *(const float4*)(zr + kb + 20);
            v16h a;
            a[0]  = (_Float16)f0.x; a[1]  = (_Float16)f0.y;
            a[2]  = (_Float16)f0.z; a[3]  = (_Float16)f0.w;
            a[4]  = (_Float16)f1.x; a[5]  = (_Float16)f1.y;
            a[6]  = (_Float16)f1.z; a[7]  = (_Float16)f1.w;
            a[8]  = (_Float16)f2.x; a[9]  = (_Float16)f2.y;
            a[10] = (_Float16)f2.z; a[11] = (_Float16)f2.w;
            a[12] = (_Float16)f3.x; a[13] = (_Float16)f3.y;
            a[14] = (_Float16)f3.z; a[15] = (_Float16)f3.w;
            A[kk] = a;
        }
    }

    float best[8];
    int   bidx[8];
    #pragma unroll
    for (int v = 0; v < 8; ++v) { best[v] = 3.4e38f; bidx[v] = 0; }

    const int crow = threadIdx.x >> 3;         // 0..15 : code row for staging
    const int cseg = (threadIdx.x & 7) * 64;   // 64-half segment within row

    // ---- Prologue: stage tile 0 into buffer 0 ----
    {
        const float4* src = (const float4*)(cbh + (size_t)crow * D_DIM + cseg);
        float4* dst = (float4*)(&tile[0][0] + crow * 520 + cseg);
        #pragma unroll
        for (int i = 0; i < 8; ++i) dst[i] = src[i];
    }
    __syncthreads();

    int p = 0;
    for (int nt = 0; nt < K_CODES / 16; ++nt) {
        const int n0 = nt * 16;

        // ---- Prefetch tile nt+1 into registers (overlaps the WMMAs).
        // For nt == 511 this reads the (valid) csq region of the workspace;
        // the staged data is never consumed.
        float4 rg[8];
        {
            const float4* src =
                (const float4*)(cbh + (size_t)(n0 + 16 + crow) * D_DIM + cseg);
            #pragma unroll
            for (int i = 0; i < 8; ++i) rg[i] = src[i];
        }
        const float cs = csq[n0 + lh];   // per-lane, L2-resident

        // ---- 16 WMMAs accumulate dot(z_row, code) over D=512.
        // B fragments pipelined one stage ahead (two live register buffers).
        const _Float16* tp = &tile[p][0];
        v8f acc = {0.f, 0.f, 0.f, 0.f, 0.f, 0.f, 0.f, 0.f};
        v16h bcur = load_bfrag(tp, lh, hi, 0);
        #pragma unroll
        for (int kk = 0; kk < 16; ++kk) {
            v16h bnext = (kk < 15) ? load_bfrag(tp, lh, hi, kk + 1) : bcur;
            acc = __builtin_amdgcn_wmma_f32_16x16x32_f16(
                      false, A[kk], false, bcur, (short)0, acc, false, false);
            bcur = bnext;
        }

        // C layout: lane (hi,lh) slot v -> row hi*8+v, column/code n0+lh
        const int myn = n0 + lh;
        #pragma unroll
        for (int v = 0; v < 8; ++v) {
            float s = __builtin_fmaf(-2.0f, acc[v], cs);
            if (s < best[v]) { best[v] = s; bidx[v] = myn; }
        }

        // ---- Store prefetched tile into the other buffer, flip. ----
        {
            float4* dst = (float4*)(&tile[p ^ 1][0] + crow * 520 + cseg);
            #pragma unroll
            for (int i = 0; i < 8; ++i) dst[i] = rg[i];
        }
        __syncthreads();
        p ^= 1;
    }

    // ---- cross-lane min within each 16-lane group (first-occurrence tiebreak)
    #pragma unroll
    for (int off = 8; off >= 1; off >>= 1) {
        #pragma unroll
        for (int v = 0; v < 8; ++v) {
            float ov = __shfl_xor(best[v], off, 32);
            int   oi = __shfl_xor(bidx[v], off, 32);
            if (ov < best[v] || (ov == best[v] && oi < bidx[v])) {
                best[v] = ov; bidx[v] = oi;
            }
        }
    }
    if (lh == 0) {
        #pragma unroll
        for (int v = 0; v < 8; ++v) {
            const int row = r0 + hi * 8 + v;
            idx_i[row] = bidx[v];
            idx_f[row] = (float)bidx[v];
        }
    }
}

// -------- Kernel 3: gather z_q (f32) + deterministic partial loss ----------
__global__ __launch_bounds__(256) void vq_gather_loss(const float* __restrict__ z,
                                                      const float* __restrict__ cb,
                                                      const int* __restrict__ idx_i,
                                                      float* __restrict__ zq_out,
                                                      double* __restrict__ part) {
    __shared__ double red[256];
    const int rbase = blockIdx.x * 64;
    double s = 0.0;
    for (int i = threadIdx.x * 4; i < 64 * D_DIM; i += 256 * 4) {
        const int r = rbase + (i >> 9);
        const int d = i & (D_DIM - 1);
        const float4 zv = *(const float4*)(z  + (size_t)r * D_DIM + d);
        const float4 qv = *(const float4*)(cb + (size_t)idx_i[r] * D_DIM + d);
        *(float4*)(zq_out + (size_t)r * D_DIM + d) = qv;
        float dx = zv.x - qv.x, dy = zv.y - qv.y;
        float dz = zv.z - qv.z, dw = zv.w - qv.w;
        s += (double)(dx * dx) + (double)(dy * dy)
           + (double)(dz * dz) + (double)(dw * dw);
    }
    red[threadIdx.x] = s;
    __syncthreads();
    for (int off = 128; off > 0; off >>= 1) {
        if (threadIdx.x < off) red[threadIdx.x] += red[threadIdx.x + off];
        __syncthreads();
    }
    if (threadIdx.x == 0) part[blockIdx.x] = red[0];
}

// -------- Kernel 4: final deterministic loss reduction ---------------------
__global__ __launch_bounds__(256) void vq_loss_final(const double* __restrict__ part,
                                                     float* __restrict__ loss_out) {
    __shared__ double red[256];
    double s = part[threadIdx.x] + part[threadIdx.x + 256];
    red[threadIdx.x] = s;
    __syncthreads();
    for (int off = 128; off > 0; off >>= 1) {
        if (threadIdx.x < off) red[threadIdx.x] += red[threadIdx.x + off];
        __syncthreads();
    }
    if (threadIdx.x == 0) {
        double mse = red[0] / (double)((size_t)B_ROWS * (size_t)D_DIM);
        *loss_out = (float)(1.25 * mse);   // codebook + BETA*commitment
    }
}

extern "C" void kernel_launch(void* const* d_in, const int* in_sizes, int n_in,
                              void* d_out, int out_size, void* d_ws, size_t ws_size,
                              hipStream_t stream) {
    const float* z  = (const float*)d_in[0];
    const float* cb = (const float*)d_in[1];

    float* out    = (float*)d_out;
    float* zq_out = out;                                  // [B, D]
    float* idx_f  = out + (size_t)B_ROWS * D_DIM;         // [B]
    float* loss   = idx_f + B_ROWS;                       // scalar

    char* ws = (char*)d_ws;
    _Float16* cbh = (_Float16*)ws;                                     // 8 MB
    float*    csq = (float*)(ws + (size_t)K_CODES * D_DIM * 2);        // 32 KB
    int*    idx_i = (int*)(ws + (size_t)K_CODES * D_DIM * 2
                              + (size_t)K_CODES * 4);                  // 128 KB
    double*  part = (double*)(ws + (size_t)K_CODES * D_DIM * 2
                                 + (size_t)K_CODES * 4
                                 + (size_t)B_ROWS * 4);                // 4 KB

    vq_prep_codebook<<<K_CODES, 128, 0, stream>>>(cb, cbh, csq);
    vq_argmin<<<B_ROWS / 64, 128, 0, stream>>>(z, cbh, csq, idx_i, idx_f);
    vq_gather_loss<<<B_ROWS / 64, 256, 0, stream>>>(z, cb, idx_i, zq_out, part);
    vq_loss_final<<<1, 256, 0, stream>>>(part, loss);
}
